// MultiHeadSelfAttention_39891656245916
// MI455X (gfx1250) — matrix operand order repaired and checked by reading, not compiled, not run on hardware
//
#include <hip/hip_runtime.h>

// ---------------------------------------------------------------------------
// MultiHeadSelfAttention for MI455X (gfx1250, wave32, WMMA)
// B=4, T=1024, DIN=512, DLIN=1024, DK=DV=1024, NH=16 (dk=dv=64)
//
// Pipeline (all f16 WMMA, f32 accumulate):
//  0) convert data / W_in[:, :512] / W_q / W_k / W_v to f16 in workspace
//  1) h   = data16 @ Win16^T + W_in[:,512+t]  (identity one-hot folded to bias)
//  2) q16 = h16 @ Wq16^T ; k16 = h16 @ Wk16^T ; vT = (h16 @ Wv16^T) transposed
//  3) flash attention per (b, head): streaming softmax, P@V via WMMA
// GEMMs double-buffer the shared B-tile in LDS via async global->LDS DMA
// (ASYNCcnt), so the 8 waves of a block do not redundantly re-load weights.
// ---------------------------------------------------------------------------

#define DEV_INLINE static __device__ __forceinline__

typedef __attribute__((ext_vector_type(16))) _Float16 v16h;
typedef __attribute__((ext_vector_type(8)))  _Float16 v8h;
typedef __attribute__((ext_vector_type(8)))  float    v8f;
typedef int vint4 __attribute__((vector_size(16)));   // builtin's b128 pointee

#if defined(__has_builtin)
#if __has_builtin(__builtin_amdgcn_global_load_async_to_lds_b128) && \
    __has_builtin(__builtin_amdgcn_s_wait_asynccnt)
#define HAVE_ASYNC_LDS 1
#endif
#endif
#ifndef HAVE_ASYNC_LDS
#define HAVE_ASYNC_LDS 0
#endif

// Copy 16 bytes global -> LDS. Async (ASYNCcnt-tracked) when available.
DEV_INLINE void copy16B_g2l(const _Float16* g, _Float16* l) {
#if HAVE_ASYNC_LDS
  __builtin_amdgcn_global_load_async_to_lds_b128(
      (__attribute__((address_space(1))) vint4*)(void*)const_cast<_Float16*>(g),
      (__attribute__((address_space(3))) vint4*)(void*)l, 0, 0);
#else
  *(v8h*)l = *(const v8h*)g;
#endif
}

DEV_INLINE void wait_async_le2() {
#if HAVE_ASYNC_LDS
  __builtin_amdgcn_s_wait_asynccnt(2);
#endif
}
DEV_INLINE void wait_async_le0() {
#if HAVE_ASYNC_LDS
  __builtin_amdgcn_s_wait_asynccnt(0);
#endif
}

// A fragment: 16x32 f16 tile, rows m = lane%16 of row-major [16, ld] matrix.
// Lanes 0-15 hold K = 0..15 (elems 0-15); lanes 16-31 hold K = 16..31
// (contiguous order, matching load_fragB).
DEV_INLINE v16h load_fragA(const _Float16* p, int ld, int k0) {
  const int lane = threadIdx.x & 31;
  const _Float16* row = p + (size_t)(lane & 15) * ld + k0 + ((lane >> 4) << 4);
  return *(const v16h*)(row);
}

// B fragment: 32x16 f16 tile, B[k][n] supplied from (N,K)-row-major storage
// (i.e. W[n][k]); lane holds column n = lane%16, 16 consecutive K values
// starting at (lane/16)*16 (contiguous k order, same as load_fragA).
DEV_INLINE v16h load_fragB(const _Float16* p, int ld, int k0) {
  const int lane = threadIdx.x & 31;
  const _Float16* row = p + (size_t)(lane & 15) * ld + k0 + ((lane >> 4) << 4);
  return *(const v16h*)(row);
}

DEV_INLINE v8f wmma16(v16h a, v16h b, v8f c) {
  return __builtin_amdgcn_wmma_f32_16x16x32_f16(false, a, false, b, (short)0, c,
                                                false, false);
}

// ------------------------------ conversions --------------------------------

__global__ void cvt_f32_to_f16(const float* __restrict__ in,
                               _Float16* __restrict__ out, int n) {
  int i = blockIdx.x * blockDim.x + threadIdx.x;
  if (i < n) out[i] = (_Float16)in[i];
}

// Extract first 512 columns of W_in [1024 x 1536] -> Win16 [1024 x 512]
__global__ void cvt_win(const float* __restrict__ W_in,
                        _Float16* __restrict__ out) {
  int i = blockIdx.x * blockDim.x + threadIdx.x;  // over 1024*512
  int n = i >> 9, c = i & 511;
  out[i] = (_Float16)W_in[(size_t)n * 1536 + c];
}

// ------------------------------ GEMM ---------------------------------------
// C[m][n] = sum_k A16[m][k] * W16[n][k]   (M x N, K inner)
// Block: 256 threads = 8 waves; block tile 128(M) x 128(N); each wave owns a
// 16-row strip across all 8 column tiles (8 x v8f accumulators). The shared
// 128x32 B-tile is staged in LDS with async DMA, double-buffered.
// mode 0: out16[m*N+n] = (f16)C                           (q, k projections)
// mode 1: C += W_in_f32[n*1536 + 512 + (m%1024)]; h16     (input projection)
// mode 2: vT[((b*16+h)*64+dv)*1024 + t] = (f16)C          (v, transposed)
__global__ __launch_bounds__(256) void gemm_awt(
    const _Float16* __restrict__ A, const _Float16* __restrict__ W,
    int N, int K,
    const float* __restrict__ bias_src,  // W_in fp32 base for mode 1
    _Float16* __restrict__ out16, int mode) {
  __shared__ __attribute__((aligned(32))) _Float16 Bs[2][128 * 32];  // 2x8KB

  const int tid = threadIdx.x;
  const int wave = tid >> 5;
  const int lane = tid & 31;
  const int col = lane & 15, half = lane >> 4;

  const int m0 = blockIdx.x * 128 + wave * 16;
  const int n0 = blockIdx.y * 128;

  v8f acc[8] = {};
  const _Float16* Abase = A + (size_t)m0 * K;

  // Cooperative stage of B tile [128 rows x 32 K] (f16) into Bs[buf]:
  // 512 chunks of 16B; 256 threads x 2 chunks (=> 2 async ops per wave).
  auto stage = [&](int buf, int k0) {
#pragma unroll
    for (int c = tid; c < 512; c += 256) {
      int rowc = c >> 2;         // 0..127
      int kc = (c & 3) << 3;     // 0, 8, 16, 24
      copy16B_g2l(W + (size_t)(n0 + rowc) * K + k0 + kc,
                  &Bs[buf][rowc * 32 + kc]);
    }
  };

  const int nk = K / 32;
  stage(0, 0);
  for (int kk = 0; kk < nk; ++kk) {
    const int buf = kk & 1;
    const bool more = (kk + 1) < nk;
    if (more) {
      stage(buf ^ 1, (kk + 1) * 32);  // prefetch next tile (other buffer)
      wait_async_le2();               // retire this tile's 2 per-wave DMAs
    } else {
      wait_async_le0();
    }
    __syncthreads();                  // all waves' DMAs for buf are visible

    v16h a = load_fragA(Abase, K, kk * 32);
#pragma unroll
    for (int nt = 0; nt < 8; ++nt) {
      v16h b = load_fragB(&Bs[buf][nt * 16 * 32], 32, 0);
      acc[nt] = wmma16(a, b, acc[nt]);
    }
    __syncthreads();                  // done reading buf before re-staging it
  }

#pragma unroll
  for (int nt = 0; nt < 8; ++nt) {
#pragma unroll
    for (int r = 0; r < 8; ++r) {
      int row = m0 + r + 8 * half;    // D layout: VGPR r -> row r + 8*half
      int c = n0 + nt * 16 + col;
      float v = acc[nt][r];
      if (mode == 1)
        v += bias_src[(size_t)c * 1536 + 512 + (row & 1023)];
      if (mode == 2) {
        int b = row >> 10, t = row & 1023;
        int h = c >> 6, dv = c & 63;
        out16[(((size_t)b * 16 + h) * 64 + dv) * 1024 + t] = (_Float16)v;
      } else {
        out16[(size_t)row * N + c] = (_Float16)v;
      }
    }
  }
}

// --------------------------- flash attention -------------------------------
// Grid: B*NH*(T/128) blocks, 256 threads (8 waves). Each wave: 16 query rows
// of one head; streams KV in blocks of 32 with online softmax; P re-layout
// D-frag -> A-frag via per-wave LDS tile; O += P @ V with B-frags from vT.
__global__ __launch_bounds__(256) void attn_kernel(
    const _Float16* __restrict__ q16, const _Float16* __restrict__ k16,
    const _Float16* __restrict__ vT, float* __restrict__ out) {
  __shared__ __attribute__((aligned(32))) _Float16 plds[8][16 * 32];

  const int wave = threadIdx.x >> 5;
  const int lane = threadIdx.x & 31;
  const int col = lane & 15, half = lane >> 4;

  const int qblk = blockIdx.x & 7;
  const int bh = blockIdx.x >> 3;
  const int b = bh >> 4, h = bh & 15;
  const int qrow0 = qblk * 128 + wave * 16;

  const _Float16* Qb = q16 + ((size_t)b * 1024 + qrow0) * 1024 + h * 64;
  const _Float16* Kb = k16 + ((size_t)b * 1024) * 1024 + h * 64;
  const _Float16* Vh = vT + (((size_t)b * 16 + h) * 64) * 1024;

  const v16h aq0 = load_fragA(Qb, 1024, 0);
  const v16h aq1 = load_fragA(Qb, 1024, 32);

  float mrow[8], lrow[8];
  v8f o[4] = {};
#pragma unroll
  for (int r = 0; r < 8; ++r) { mrow[r] = -__builtin_inff(); lrow[r] = 0.f; }

  const float scale = 0.125f;  // 1/sqrt(64)
  _Float16* P = &plds[wave][0];

  for (int jb = 0; jb < 32; ++jb) {  // 32 KV-blocks of 32 -> T = 1024
    const _Float16* Kt = Kb + (size_t)(jb * 32) * 1024;
    v8f s0 = {}, s1 = {};
    s0 = wmma16(aq0, load_fragB(Kt, 1024, 0), s0);
    s0 = wmma16(aq1, load_fragB(Kt, 1024, 32), s0);
    s1 = wmma16(aq0, load_fragB(Kt + (size_t)16 * 1024, 1024, 0), s1);
    s1 = wmma16(aq1, load_fragB(Kt + (size_t)16 * 1024, 1024, 32), s1);

#pragma unroll
    for (int r = 0; r < 8; ++r) {
      float v0 = s0[r] * scale;
      float v1 = s1[r] * scale;
      float tmax = fmaxf(v0, v1);
#pragma unroll
      for (int off = 8; off; off >>= 1)
        tmax = fmaxf(tmax, __shfl_xor(tmax, off));       // reduce 16-lane group
      float newm = fmaxf(mrow[r], tmax);
      float p0 = __expf(v0 - newm);
      float p1 = __expf(v1 - newm);
      float psum = p0 + p1;
#pragma unroll
      for (int off = 8; off; off >>= 1) psum += __shfl_xor(psum, off);
      float corr = __expf(mrow[r] - newm);
      lrow[r] = lrow[r] * corr + psum;
      mrow[r] = newm;
#pragma unroll
      for (int ct = 0; ct < 4; ++ct) o[ct][r] *= corr;
      P[(r + 8 * half) * 32 + col] = (_Float16)p0;
      P[(r + 8 * half) * 32 + 16 + col] = (_Float16)p1;
    }

    v16h pa = load_fragA(P, 32, 0);  // same-wave LDS RAW; compiler waits dscnt
#pragma unroll
    for (int ct = 0; ct < 4; ++ct) {
      v16h bv = load_fragB(Vh + (size_t)(ct * 16) * 1024 + jb * 32, 1024, 0);
      o[ct] = wmma16(pa, bv, o[ct]);
    }
  }

#pragma unroll
  for (int ct = 0; ct < 4; ++ct) {
#pragma unroll
    for (int r = 0; r < 8; ++r) {
      int row = qrow0 + r + 8 * half;
      float val = o[ct][r] / lrow[r];
      out[((size_t)b * 1024 + row) * 1024 + h * 64 + ct * 16 + col] = val;
    }
  }
}

// ------------------------------ launcher -----------------------------------

extern "C" void kernel_launch(void* const* d_in, const int* in_sizes, int n_in,
                              void* d_out, int out_size, void* d_ws,
                              size_t ws_size, hipStream_t stream) {
  const float* data = (const float*)d_in[0];  // [4,1024,512]
  const float* W_in = (const float*)d_in[1];  // [1024,1536]
  const float* W_q  = (const float*)d_in[2];  // [1024,1024]
  const float* W_k  = (const float*)d_in[3];
  const float* W_v  = (const float*)d_in[4];
  float* out = (float*)d_out;                 // [4,1024,1024] f32

  char* ws = (char*)d_ws;
  size_t off = 0;
  auto alloc16 = [&](size_t elems) -> _Float16* {
    _Float16* p = (_Float16*)(ws + off);
    off = (off + elems * sizeof(_Float16) + 255) & ~(size_t)255;
    return p;
  };
  _Float16* data16 = alloc16((size_t)4096 * 512);   // 4 MB
  _Float16* Win16  = alloc16((size_t)1024 * 512);   // 1 MB
  _Float16* Wq16   = alloc16((size_t)1024 * 1024);  // 2 MB
  _Float16* Wk16   = alloc16((size_t)1024 * 1024);
  _Float16* Wv16   = alloc16((size_t)1024 * 1024);
  _Float16* h16    = alloc16((size_t)4096 * 1024);  // 8 MB
  _Float16* q16    = alloc16((size_t)4096 * 1024);
  _Float16* k16    = alloc16((size_t)4096 * 1024);
  _Float16* vT     = alloc16((size_t)4096 * 1024);  // [B,H,64,T]
  (void)ws_size; (void)in_sizes; (void)n_in; (void)out_size;

  // Stage 0: conversions
  cvt_f32_to_f16<<<(4096 * 512) / 256, 256, 0, stream>>>(data, data16,
                                                         4096 * 512);
  cvt_win<<<(1024 * 512) / 256, 256, 0, stream>>>(W_in, Win16);
  cvt_f32_to_f16<<<(1024 * 1024) / 256, 256, 0, stream>>>(W_q, Wq16,
                                                          1024 * 1024);
  cvt_f32_to_f16<<<(1024 * 1024) / 256, 256, 0, stream>>>(W_k, Wk16,
                                                          1024 * 1024);
  cvt_f32_to_f16<<<(1024 * 1024) / 256, 256, 0, stream>>>(W_v, Wv16,
                                                          1024 * 1024);

  // Stage 1: h = data @ Win^T + positional bias   (M=4096, N=1024, K=512)
  gemm_awt<<<dim3(32, 8), 256, 0, stream>>>(data16, Win16, 1024, 512, W_in,
                                            h16, /*mode=*/1);

  // Stage 2: projections (M=4096, N=1024, K=1024)
  gemm_awt<<<dim3(32, 8), 256, 0, stream>>>(h16, Wq16, 1024, 1024, nullptr,
                                            q16, /*mode=*/0);
  gemm_awt<<<dim3(32, 8), 256, 0, stream>>>(h16, Wk16, 1024, 1024, nullptr,
                                            k16, /*mode=*/0);
  gemm_awt<<<dim3(32, 8), 256, 0, stream>>>(h16, Wv16, 1024, 1024, nullptr,
                                            vT, /*mode=*/2);

  // Stage 3: flash attention; B*NH*(T/128) = 512 blocks
  attn_kernel<<<512, 256, 0, stream>>>(q16, k16, vT, out);
}
